// Encoder_15745350107679
// MI455X (gfx1250) — matrix operand order repaired and checked by reading, not compiled, not run on hardware
//
#include <hip/hip_runtime.h>
#include <stdint.h>

// Problem constants (from reference): V=32000, E=256, H=256, B=64, T=512
#define E_  256
#define H_  256
#define B_  64
#define T_  512
#define G4H 1024   // 4*H

// ---- vector types matching the gfx1250 WMMA / TDM builtins ----
typedef __attribute__((ext_vector_type(16))) __bf16 v16bf;
typedef __attribute__((ext_vector_type(8)))  __bf16 v8bf;
typedef __attribute__((ext_vector_type(8)))  float  v8f;
typedef __attribute__((ext_vector_type(4)))  unsigned int u32x4;
typedef __attribute__((ext_vector_type(8)))  int i32x8;
typedef __attribute__((ext_vector_type(4)))  int i32x4;

union BF16Frag { v16bf v; v8bf h[2]; };

#if defined(__has_builtin)
#if __has_builtin(__builtin_amdgcn_tensor_load_to_lds) && __has_builtin(__builtin_amdgcn_s_wait_tensorcnt)
#define HAS_TDM 1
#endif
#endif
#ifndef HAS_TDM
#define HAS_TDM 0
#endif

__device__ __forceinline__ __bf16 f2bf(float f) {
  unsigned u = __builtin_bit_cast(unsigned, f);
  unsigned r = (u + 0x7FFFu + ((u >> 16) & 1u)) >> 16;   // round-to-nearest-even
  unsigned short s = (unsigned short)r;
  return __builtin_bit_cast(__bf16, s);
}
__device__ __forceinline__ float bf2f(__bf16 b) {
  unsigned short s = __builtin_bit_cast(unsigned short, b);
  return __builtin_bit_cast(float, ((unsigned)s) << 16);
}
__device__ __forceinline__ float sigmf(float x) { return 1.0f / (1.0f + __expf(-x)); }
__device__ __forceinline__ float tanh_fast(float x) {
  x = fminf(fmaxf(x, -15.0f), 15.0f);
  float e = __expf(2.0f * x);
  return (e - 1.0f) / (e + 1.0f);
}

// ------------------------------------------------------------------
// Prep: swizzle W_ih / W_hh (f32 -> bf16) into WMMA B-fragment order.
// B fragment (32Kx16N bf16): lane l = N-col l%16, 16 halfs = K {8h..8h+7, 16+8h..16+8h+7}, h = l/16.
// w_ih frags: frag = ntile*8 + kt, ntile over 2 dirs * 1024 rows / 16 = 128.
// w_hh frags: frag = ((dir*16 + wave)*4 + gate)*8 + kt; wave owns h-cols [16w,16w+16).
// ------------------------------------------------------------------
__global__ void prep_kernel(const float* __restrict__ wih_f, const float* __restrict__ whh_f,
                            const float* __restrict__ bih_f, const float* __restrict__ bhh_f,
                            const float* __restrict__ wih_b, const float* __restrict__ whh_b,
                            const float* __restrict__ bih_b, const float* __restrict__ bhh_b,
                            __bf16* __restrict__ wih_frag, __bf16* __restrict__ whh_frag,
                            float* __restrict__ bias)
{
  int tid  = blockIdx.x * blockDim.x + threadIdx.x;   // 0 .. 524287
  int frag = tid >> 9;                                // 512 elements per fragment
  int lane = (tid >> 4) & 31;
  int j    = tid & 15;

  { // W_ih (gates = xe @ W_ih^T  =>  B[k][n] = W_ih[n][k])
    int ntile = frag >> 3, kt = frag & 7;
    int n_global = ntile * 16 + (lane & 15);
    int dir = n_global >> 10;
    int row = n_global & 1023;
    int k   = kt * 32 + ((j & 8) << 1) + (lane >> 4) * 8 + (j & 7);
    const float* w = dir ? wih_b : wih_f;
    wih_frag[(size_t)frag * 512 + lane * 16 + j] = f2bf(w[(size_t)row * E_ + k]);
  }
  { // W_hh
    int f2 = frag;
    int kt = f2 & 7;  f2 >>= 3;
    int g  = f2 & 3;  f2 >>= 2;
    int wv = f2 & 15; f2 >>= 4;
    int dir = f2 & 1;
    int row = g * H_ + wv * 16 + (lane & 15);
    int k   = kt * 32 + ((j & 8) << 1) + (lane >> 4) * 8 + (j & 7);
    const float* w = dir ? whh_b : whh_f;
    whh_frag[(size_t)frag * 512 + lane * 16 + j] = f2bf(w[(size_t)row * H_ + k]);
  }
  if (tid < 2048) {
    int dir = tid >> 10, n = tid & 1023;
    bias[tid] = dir ? (bih_b[n] + bhh_b[n]) : (bih_f[n] + bhh_f[n]);
  }
}

// ------------------------------------------------------------------
// Embedding gather -> bf16 activation matrix xe[B*T][E]
// ------------------------------------------------------------------
__global__ void gather_kernel(const int* __restrict__ x, const float* __restrict__ embed,
                              __bf16* __restrict__ xe)
{
  int m = blockIdx.x;            // b*T + t
  int e = threadIdx.x;           // 0..255
  int idx = x[m];
  xe[(size_t)m * E_ + e] = f2bf(embed[(size_t)idx * E_ + e]);
}

// ------------------------------------------------------------------
// xproj GEMM:  xproj_bf16[dir][t][b][4H] = xe @ W_ih^T + (b_ih+b_hh)
// One wave = 16(M) x 64(N) strip. Whole A strip preloaded into VGPRs;
// B fragments double-buffered so loads for kt+1 are in flight while the
// WMMAs for kt execute (breaks the load->wait0->wmma serialization).
// ------------------------------------------------------------------
__global__ void xproj_kernel(const __bf16* __restrict__ xe, const __bf16* __restrict__ wih_frag,
                             const float* __restrict__ bias, __bf16* __restrict__ xproj)
{
  const int lane   = threadIdx.x & 31;
  const int wv     = threadIdx.x >> 5;          // 0..7
  const int gw     = blockIdx.x * 8 + wv;       // 0..65535
  const int mtile  = gw >> 5;                   // 0..2047 (M = B*T = 32768)
  const int nchunk = gw & 31;                   // 0..31   (4 ntiles each, N = 2048)
  const int hi  = lane >> 4;
  const int nlo = lane & 15;

  // Preload entire 16x256 A strip (A frag: lanes 0-15 rows, K {0..7,16..23};
  // lanes 16-31 same rows, K {8..15,24..31})
  const __bf16* arow = xe + (size_t)(mtile * 16 + (lane & 15)) * E_;
  BF16Frag a[8];
  #pragma unroll
  for (int kt = 0; kt < 8; ++kt) {
    a[kt].h[0] = *(const v8bf*)(arow + kt * 32 + hi * 8);
    a[kt].h[1] = *(const v8bf*)(arow + kt * 32 + hi * 8 + 16);
  }

  v8f acc[4];
  #pragma unroll
  for (int nt = 0; nt < 4; ++nt) {
    float bv = bias[(nchunk * 4 + nt) * 16 + nlo];
    #pragma unroll
    for (int r = 0; r < 8; ++r) acc[nt][r] = bv;
  }

  const __bf16* bp = wih_frag + (size_t)(nchunk * 4) * 8 * 512 + lane * 16;
  v16bf b[2][4];
  #pragma unroll
  for (int nt = 0; nt < 4; ++nt) b[0][nt] = *(const v16bf*)(bp + (size_t)nt * 8 * 512);

  #pragma unroll
  for (int kt = 0; kt < 8; ++kt) {
    const int cur = kt & 1, nxt = cur ^ 1;
    if (kt < 7) {
      #pragma unroll
      for (int nt = 0; nt < 4; ++nt)
        b[nxt][nt] = *(const v16bf*)(bp + (size_t)nt * 8 * 512 + (kt + 1) * 512);
    }
    #pragma unroll
    for (int nt = 0; nt < 4; ++nt)
      acc[nt] = __builtin_amdgcn_wmma_f32_16x16x32_bf16(false, a[kt].v, false, b[cur][nt],
                                                        (short)0, acc[nt], false, false);
  }

  // C/D layout: VGPR r, lane l -> row (r + 8*(l>>4)), col (l&15)
  #pragma unroll
  for (int nt = 0; nt < 4; ++nt) {
    int n_global = (nchunk * 4 + nt) * 16 + nlo;
    int dir = n_global >> 10;
    int g   = n_global & 1023;
    #pragma unroll
    for (int r = 0; r < 8; ++r) {
      int m = mtile * 16 + r + 8 * hi;
      int bb = m >> 9;      // m = b*T + t
      int t  = m & 511;
      xproj[(((size_t)dir * T_ + t) * B_ + bb) * G4H + g] = f2bf(acc[nt][r]);
    }
  }
}

// ------------------------------------------------------------------
// Recurrent scan. grid = 8 independent WGs: dir {0,1} x batch-quarter {0..3}.
// Block = 512 threads = 16 waves; wave w owns h-columns [16w,16w+16) and its
// four gate tiles -> gate fusion is wave-local. h lives in LDS (bf16 A-matrix);
// c stays in the f32 accumulator register layout for the whole scan.
// The per-step xproj slab (16 x 4H bf16 = 32 KB, contiguous) is DMA'd into LDS
// by the Tensor Data Mover, overlapping the previous step's elementwise tail.
// ------------------------------------------------------------------
__global__ void lstm_kernel(const __bf16* __restrict__ xproj, const __bf16* __restrict__ whh_frag,
                            const int* __restrict__ lengths, float* __restrict__ out)
{
  const int dir   = blockIdx.x >> 2;
  const int bbase = (blockIdx.x & 3) * 16;
  const int lane  = threadIdx.x & 31;
  const int wv    = threadIdx.x >> 5;     // 0..15
  const int hi  = lane >> 4;
  const int nlo = lane & 15;

  __shared__ __bf16 hbuf[16][264];        // +8 halfs pad -> conflict-free A loads
  __shared__ __bf16 xslab[16 * G4H];      // 32 KiB staging for one step's xproj rows

  for (int i = threadIdx.x; i < 16 * 264; i += blockDim.x)
    ((__bf16*)hbuf)[i] = __builtin_bit_cast(__bf16, (unsigned short)0);

  v8f c, h;
  int len8[8];
  #pragma unroll
  for (int r = 0; r < 8; ++r) { c[r] = 0.0f; h[r] = 0.0f; len8[r] = lengths[bbase + r + 8 * hi]; }

#if HAS_TDM
  // 1-D TDM descriptor: copy 16384 bf16 elements (16 rows x 4H, contiguous)
  // from xproj[dir][t][bbase][0] into xslab.
  auto tdm_fetch = [&](int t) {
    unsigned long long ga =
        (unsigned long long)(size_t)(xproj + (((size_t)dir * T_ + t) * B_ + bbase) * G4H);
    unsigned lds = (unsigned)(size_t)(&xslab[0]);     // flat addr low 32 = LDS byte addr
    u32x4 g0 = { 1u,                                  // count=1, user descriptor
                 lds,                                  // lds_addr
                 (unsigned)(ga & 0xFFFFFFFFu),         // global_addr[31:0]
                 (unsigned)((ga >> 32) & 0x01FFFFFFu) | (2u << 30) };  // addr[56:32] | type=2
    i32x8 g1 = { (int)(1u << 16),                     // wg_mask=0, data_size=1 (2 bytes)
                 (int)((16384u & 0xFFFFu) << 16),     // tensor_dim0[15:0]
                 (int)(1u << 16),                     // tensor_dim0[31:16]=0 | tensor_dim1[15:0]=1
                 (int)(16384u << 16),                 // tensor_dim1 hi=0 | tile_dim0=16384
                 1,                                   // tile_dim1=1, tile_dim2=0
                 16384,                               // tensor_dim0_stride[31:0]
                 0, 0 };
    i32x4 z4 = { 0, 0, 0, 0 };
    i32x8 z8 = { 0, 0, 0, 0, 0, 0, 0, 0 };
    // 6-arg flavor (clang-23 / therock headers): groups 2,3 + aux group, then cpol
    __builtin_amdgcn_tensor_load_to_lds(g0, g1, z4, z4, z8, 0);
  };
  if (wv == 0) tdm_fetch(dir ? (T_ - 1) : 0);
#endif
  __syncthreads();

  const __bf16* bp0 =
      whh_frag + (((size_t)dir * 16 + wv) * 4) * 8 * 512 + lane * 16;  // gate g: +g*8*512
  const size_t out_ch = (size_t)dir * H_ + wv * 16 + nlo;

  for (int step = 0; step < T_; ++step) {
    const int t = dir ? (T_ - 1 - step) : step;

    v8f acc_i, acc_f, acc_g, acc_o;
#if HAS_TDM
    if (wv == 0) __builtin_amdgcn_s_wait_tensorcnt(0);
    __syncthreads();                          // slab for this step visible to all waves
    const int col = wv * 16 + nlo;
    #pragma unroll
    for (int r = 0; r < 8; ++r) {
      size_t ro = (size_t)(r + 8 * hi) * G4H + col;
      acc_i[r] = bf2f(xslab[ro + 0 * H_]);
      acc_f[r] = bf2f(xslab[ro + 1 * H_]);
      acc_g[r] = bf2f(xslab[ro + 2 * H_]);
      acc_o[r] = bf2f(xslab[ro + 3 * H_]);
    }
#else
    const size_t xbase = (((size_t)dir * T_ + t) * B_ + bbase) * G4H + wv * 16 + nlo;
    #pragma unroll
    for (int r = 0; r < 8; ++r) {
      size_t ro = (size_t)(r + 8 * hi) * G4H;
      acc_i[r] = bf2f(xproj[xbase + ro + 0 * H_]);
      acc_f[r] = bf2f(xproj[xbase + ro + 1 * H_]);
      acc_g[r] = bf2f(xproj[xbase + ro + 2 * H_]);
      acc_o[r] = bf2f(xproj[xbase + ro + 3 * H_]);
    }
    if (step + 1 < T_) {
      int tn = dir ? (T_ - 2 - step) : (step + 1);
      __builtin_prefetch(&xproj[(((size_t)dir * T_ + tn) * B_ + bbase) * G4H], 0, 1);
    }
#endif

    // gates += h @ W_hh^T : 8 k-steps x 4 gate tiles, double-buffered fragments
    const __bf16* ap0 = &hbuf[lane & 15][hi * 8];
    BF16Frag af[2];
    v16bf bb[2][4];
    af[0].h[0] = *(const v8bf*)(ap0);
    af[0].h[1] = *(const v8bf*)(ap0 + 16);
    #pragma unroll
    for (int g = 0; g < 4; ++g) bb[0][g] = *(const v16bf*)(bp0 + (size_t)g * 8 * 512);

    #pragma unroll
    for (int kt = 0; kt < 8; ++kt) {
      const int cur = kt & 1, nxt = cur ^ 1;
      if (kt < 7) {
        const __bf16* ap = ap0 + (kt + 1) * 32;
        af[nxt].h[0] = *(const v8bf*)(ap);
        af[nxt].h[1] = *(const v8bf*)(ap + 16);
        #pragma unroll
        for (int g = 0; g < 4; ++g)
          bb[nxt][g] = *(const v16bf*)(bp0 + (size_t)g * 8 * 512 + (kt + 1) * 512);
      }
      acc_i = __builtin_amdgcn_wmma_f32_16x16x32_bf16(false, af[cur].v, false, bb[cur][0], (short)0, acc_i, false, false);
      acc_f = __builtin_amdgcn_wmma_f32_16x16x32_bf16(false, af[cur].v, false, bb[cur][1], (short)0, acc_f, false, false);
      acc_g = __builtin_amdgcn_wmma_f32_16x16x32_bf16(false, af[cur].v, false, bb[cur][2], (short)0, acc_g, false, false);
      acc_o = __builtin_amdgcn_wmma_f32_16x16x32_bf16(false, af[cur].v, false, bb[cur][3], (short)0, acc_o, false, false);
    }
    __syncthreads();                    // all waves done reading hbuf (and xslab)

#if HAS_TDM
    if (wv == 0 && step + 1 < T_) tdm_fetch(dir ? (T_ - 2 - step) : (step + 1));
#endif

    // fused gates -> state update (pytorch order i,f,g,o), length masking
    #pragma unroll
    for (int r = 0; r < 8; ++r) {
      float iv = sigmf(acc_i[r]);
      float fv = sigmf(acc_f[r]);
      float gv = tanh_fast(acc_g[r]);
      float ov = sigmf(acc_o[r]);
      float cn = fv * c[r] + iv * gv;
      float hn = ov * tanh_fast(cn);
      bool valid = (t < len8[r]);
      if (valid) { c[r] = cn; h[r] = hn; }
      int b = bbase + r + 8 * hi;
      out[((size_t)b * T_ + t) * (2 * H_) + out_ch] = valid ? hn : 0.0f;
      hbuf[r + 8 * hi][wv * 16 + nlo] = f2bf(h[r]);   // masked h for next step
    }
    __syncthreads();
  }

  // final hidden / cell, laid out after output in d_out
  const size_t hid_off  = (size_t)B_ * T_ * 2 * H_;
  const size_t cell_off = hid_off + (size_t)B_ * 2 * H_;
  #pragma unroll
  for (int r = 0; r < 8; ++r) {
    int b = bbase + r + 8 * hi;
    size_t idx = (size_t)b * (2 * H_) + out_ch;
    out[hid_off  + idx] = h[r];
    out[cell_off + idx] = c[r];
  }
}

// ------------------------------------------------------------------
extern "C" void kernel_launch(void* const* d_in, const int* in_sizes, int n_in,
                              void* d_out, int out_size, void* d_ws, size_t ws_size,
                              hipStream_t stream)
{
  const int*   x     = (const int*)  d_in[0];
  const int*   len   = (const int*)  d_in[1];
  const float* embed = (const float*)d_in[2];
  const float* wih_f = (const float*)d_in[3];
  const float* whh_f = (const float*)d_in[4];
  const float* bih_f = (const float*)d_in[5];
  const float* bhh_f = (const float*)d_in[6];
  const float* wih_b = (const float*)d_in[7];
  const float* whh_b = (const float*)d_in[8];
  const float* bih_b = (const float*)d_in[9];
  const float* bhh_b = (const float*)d_in[10];
  float* out = (float*)d_out;

  // workspace layout (~146 MiB): xe_bf16 | wih_frag | whh_frag | bias | xproj_bf16
  char* ws = (char*)d_ws;
  size_t off = 0;
  __bf16* xe       = (__bf16*)(ws + off); off += (size_t)B_ * T_ * E_ * 2;       // 16 MiB
  __bf16* wih_frag = (__bf16*)(ws + off); off += (size_t)1024 * 512 * 2;         // 1 MiB
  __bf16* whh_frag = (__bf16*)(ws + off); off += (size_t)1024 * 512 * 2;         // 1 MiB
  float*  bias     = (float*) (ws + off); off += (size_t)2048 * sizeof(float);
  __bf16* xproj    = (__bf16*)(ws + off);                                        // 128 MiB

  prep_kernel<<<2048, 256, 0, stream>>>(wih_f, whh_f, bih_f, bhh_f,
                                        wih_b, whh_b, bih_b, bhh_b,
                                        wih_frag, whh_frag, bias);
  gather_kernel<<<B_ * T_, 256, 0, stream>>>(x, embed, xe);
  xproj_kernel<<<8192, 256, 0, stream>>>(xe, wih_frag, bias, xproj);
  lstm_kernel<<<8, 512, 0, stream>>>(xproj, whh_frag, len, out);
}
